// Mamba_32074815766760
// MI455X (gfx1250) — matrix operand order repaired and checked by reading, not compiled, not run on hardware
//
#include <hip/hip_runtime.h>
#include <math.h>

typedef __attribute__((ext_vector_type(16))) _Float16 v16h;
typedef __attribute__((ext_vector_type(8)))  float    v8f;
typedef unsigned int u32x4 __attribute__((ext_vector_type(4)));
typedef int          i32x4 __attribute__((ext_vector_type(4)));
typedef int          i32x8 __attribute__((ext_vector_type(8)));

#define BB    2
#define LL    512
#define HH    128
#define EHH   256
#define MM    (BB * LL)       // 1024 rows
#define KCONV 4
#define DTR   8
#define NPAD  272             // 264 padded up to multiple of 16
#define TC    16              // timesteps staged per TDM chunk

#if defined(__has_builtin)
#if __has_builtin(__builtin_amdgcn_tensor_load_to_lds)
#define HAVE_TDM 1
#endif
#if __has_builtin(__builtin_amdgcn_s_wait_tensorcnt)
#define HAVE_TCNT 1
#endif
#endif

__device__ __forceinline__ float siluf(float x) { return x / (1.0f + expf(-x)); }

// ---------------------------------------------------------------------------
// f32 -> f16 weight conversion
// ---------------------------------------------------------------------------
__global__ void k_convert(const float* __restrict__ src, _Float16* __restrict__ dst, int n) {
    int i = blockIdx.x * blockDim.x + threadIdx.x;
    if (i < n) dst[i] = (_Float16)src[i];
}

__global__ void k_convert_pad(const float* __restrict__ src, _Float16* __restrict__ dst,
                              int nsrc, int ndst, int k) {
    int i = blockIdx.x * blockDim.x + threadIdx.x;
    if (i >= ndst * k) return;
    int row = i / k;
    dst[i] = (row < nsrc) ? (_Float16)src[i] : (_Float16)0.0f;
}

// ---------------------------------------------------------------------------
// RMSNorm: one wave32 per row of H=128 (4 elems/lane), optional f16/f32 out
// ---------------------------------------------------------------------------
__global__ void k_rmsnorm(const float* __restrict__ x, const float* __restrict__ w,
                          _Float16* __restrict__ oh, float* __restrict__ of, int M) {
    int wave = (blockIdx.x * blockDim.x + threadIdx.x) >> 5;
    int lane = threadIdx.x & 31;
    if (wave >= M) return;
    const float* row = x + (size_t)wave * HH;
    float v[4], s = 0.0f;
#pragma unroll
    for (int j = 0; j < 4; ++j) { v[j] = row[lane + 32 * j]; s += v[j] * v[j]; }
#pragma unroll
    for (int off = 16; off; off >>= 1) s += __shfl_xor(s, off, 32);
    float r = rsqrtf(s * (1.0f / HH) + 1e-5f);
#pragma unroll
    for (int j = 0; j < 4; ++j) {
        int n = lane + 32 * j;
        float o = v[j] * r * w[n];
        if (oh) oh[(size_t)wave * HH + n] = (_Float16)o;
        if (of) of[(size_t)wave * HH + n] = o;
    }
}

// ---------------------------------------------------------------------------
// WMMA GEMM:  C[M,N](f32) = A[M,K](f16 rm) * W[N,K](f16 rm)^T + bias
// One wave computes a 16x(16*NBLK) strip: A fragment loaded once per k-step
// and reused across NBLK B tiles. K is a template constant -> fully unrolled.
// Fragment layouts per CDNA5 ISA 7.12.2.
// ---------------------------------------------------------------------------
template <int K, int NBLK>
__global__ void k_gemm_wmma(const _Float16* __restrict__ A, const _Float16* __restrict__ W,
                            const float* __restrict__ bias, float* __restrict__ C,
                            int M, int N, int ldc) {
    int wave = (blockIdx.x * blockDim.x + threadIdx.x) >> 5;
    int lane = threadIdx.x & 31;
    int groups_n = (N >> 4) / NBLK;
    int groups   = (M >> 4) * groups_n;
    if (wave >= groups) return;
    int tm16 = (wave / groups_n) << 4;
    int tn0  = (wave % groups_n) * NBLK;

    const int hi   = (lane >= 16);
    const int arow = tm16 + (lane & 15);

    v8f acc[NBLK];
    v8f zero = {};
#pragma unroll
    for (int nb = 0; nb < NBLK; ++nb) acc[nb] = zero;

#pragma unroll
    for (int k0 = 0; k0 < K; k0 += 32) {
        v16h a;
#pragma unroll
        for (int i = 0; i < 16; ++i) {
            // A 16x32 f16: VGPR v half p -> K = {2v+p | 16+2(v-4)+p}, lanes>=16 add 8
            int v = i >> 1, p = i & 1;
            int ka = ((v < 4) ? (2 * v + p) : (16 + 2 * (v - 4) + p)) + (hi ? 8 : 0);
            a[i] = A[(size_t)arow * K + k0 + ka];
        }
#pragma unroll
        for (int nb = 0; nb < NBLK; ++nb) {
            int bcol = ((tn0 + nb) << 4) + (lane & 15);
            v16h bf;
#pragma unroll
            for (int i = 0; i < 16; ++i) {
                // B 32x16 f16: lane = N col, half i -> K = {i | 16+(i-8)}, lanes>=16 add 8
                bf[i] = W[(size_t)bcol * K + k0 + ((i < 8) ? i : (i + 8)) + (hi ? 8 : 0)];
            }
            acc[nb] = __builtin_amdgcn_wmma_f32_16x16x32_f16(false, a, false, bf, (short)0,
                                                             acc[nb], false, false);
        }
    }
    // C/D: lane = N col (0-15), VGPR r -> M = r (+8 for lanes 16-31)
    int mo = hi ? 8 : 0;
#pragma unroll
    for (int nb = 0; nb < NBLK; ++nb) {
        int n  = ((tn0 + nb) << 4) + (lane & 15);
        float bv = bias ? bias[n] : 0.0f;
#pragma unroll
        for (int r = 0; r < 8; ++r)
            C[(size_t)(tm16 + mo + r) * ldc + n] = acc[nb][r] + bv;
    }
}

// ---------------------------------------------------------------------------
// Causal depthwise conv1d (K=4) + bias + SiLU; u comes from xs[:, 0:EH]
// ---------------------------------------------------------------------------
__global__ void k_conv_silu(const float* __restrict__ xs, const float* __restrict__ cw,
                            const float* __restrict__ cb, float* __restrict__ u,
                            _Float16* __restrict__ uh) {
    int idx = blockIdx.x * blockDim.x + threadIdx.x;
    if (idx >= MM * EHH) return;
    int m = idx / EHH, c = idx % EHH;
    int l = m % LL;
    float s = cb[c];
#pragma unroll
    for (int j = 0; j < KCONV; ++j) {
        int li = l + j - (KCONV - 1);
        if (li >= 0) s += cw[c * KCONV + j] * xs[(size_t)(m + j - (KCONV - 1)) * (2 * EHH) + c];
    }
    float o = siluf(s);
    u[idx]  = o;
    uh[idx] = (_Float16)o;
}

// ---------------------------------------------------------------------------
// delta = softplus(dbc[:, :8] @ dt_w^T + dt_b) / sqrt(EH)    (K=8 -> plain VALU)
// ---------------------------------------------------------------------------
__global__ void k_delta(const float* __restrict__ dbc, const float* __restrict__ dtw,
                        const float* __restrict__ dtb, float* __restrict__ delta) {
    int idx = blockIdx.x * blockDim.x + threadIdx.x;
    if (idx >= MM * EHH) return;
    int m = idx / EHH, c = idx % EHH;
    float s = dtb[c];
#pragma unroll
    for (int r = 0; r < DTR; ++r) s += dbc[(size_t)m * NPAD + r] * dtw[c * DTR + r];
    float sp = (s > 20.0f) ? s : log1pf(expf(s));
    delta[idx] = sp * (1.0f / 16.0f);   // 1/sqrt(256)
}

// ---------------------------------------------------------------------------
// TDM helper: 2D tile (rows x tile_w f32 elements, row stride in elements)
// from global memory into LDS at lds_off. D# layout per CDNA5 ISA.
// ---------------------------------------------------------------------------
#if defined(HAVE_TDM)
__device__ __forceinline__ void tdm_load_2d(unsigned int lds_off, const void* gptr,
                                            int tile_w, int rows, int stride_elems) {
    unsigned long long ga = (unsigned long long)gptr;
    u32x4 g0;
    g0[0] = 1u;                                            // count=1, user descriptor
    g0[1] = lds_off;                                       // lds_addr (bytes)
    g0[2] = (unsigned int)(ga & 0xffffffffu);              // global_addr[31:0]
    g0[3] = (unsigned int)((ga >> 32) & 0x1ffffffu) | (2u << 30); // [56:32] | type=2
    i32x8 g1;
    g1[0] = (int)(2u << 16);                               // data_size=4B, wg_mask=0
    g1[1] = (int)(((unsigned)tile_w & 0xffffu) << 16);     // tensor_dim0[15:0]
    g1[2] = (int)((((unsigned)tile_w >> 16) & 0xffffu) |
                  (((unsigned)rows & 0xffffu) << 16));     // tensor_dim0[31:16] | tensor_dim1[15:0]
    g1[3] = (int)((((unsigned)rows >> 16) & 0xffffu) |
                  (((unsigned)tile_w & 0xffffu) << 16));   // tensor_dim1[31:16] | tile_dim0
    g1[4] = (int)((unsigned)rows & 0xffffu);               // tile_dim1 (tile_dim2=0)
    g1[5] = (int)stride_elems;                             // tensor_dim0_stride[31:0]
    g1[6] = 0;                                             // stride hi / dim1_stride lo
    g1[7] = 0;
    i32x4 z4 = {};
#if __clang_major__ >= 23
    i32x8 z8 = {};
    __builtin_amdgcn_tensor_load_to_lds(g0, g1, z4, z4, z8, 0);
#else
    __builtin_amdgcn_tensor_load_to_lds(g0, g1, z4, z4, 0);
#endif
}
#endif

// ---------------------------------------------------------------------------
// SSM scan. Block = 8 waves = 8 channels of the same batch; all waves share the
// same B_t/C_t rows (depend on (b,t) only), so each block stages TC timesteps of
// dbc (TC x 256 f32) into LDS once via the Tensor Data Mover, then 8 waves
// consume broadcast rows. State: 128 per channel, 4 regs/lane, sequential in t.
// ---------------------------------------------------------------------------
__global__ void k_scan(const float* __restrict__ delta, const float* __restrict__ u,
                       const float* __restrict__ dbc, const float* __restrict__ Alog,
                       const float* __restrict__ Dv, float* __restrict__ y) {
    __shared__ float sBC[TC * 256];
    int wv   = threadIdx.x >> 5;
    int lane = threadIdx.x & 31;
    int wid  = blockIdx.x * 8 + wv;
    int b = wid / EHH, c = wid % EHH;

    float an[4], h[4];
#pragma unroll
    for (int j = 0; j < 4; ++j) {
        an[j] = -expf(Alog[(size_t)c * HH + lane + 32 * j]);
        h[j]  = 0.0f;
    }
    float Dc = Dv[c];

    for (int t0 = 0; t0 < LL; t0 += TC) {
        int m0 = b * LL + t0;
        // stage TC rows of [B|C] (cols 8..263 of dbc) into LDS
#if defined(HAVE_TDM)
        if (threadIdx.x < 32) {   // wave 0 issues the DMA
            tdm_load_2d((unsigned int)(unsigned long long)&sBC[0],
                        dbc + (size_t)m0 * NPAD + DTR, 256, TC, NPAD);
#if defined(HAVE_TCNT)
            __builtin_amdgcn_s_wait_tensorcnt(0);
#endif
        }
#else
        for (int i = threadIdx.x; i < TC * 256; i += 256) {
            int row = i >> 8, col = i & 255;
            sBC[i] = dbc[(size_t)(m0 + row) * NPAD + DTR + col];
        }
#endif
        __syncthreads();

        for (int tt = 0; tt < TC; ++tt) {
            int m = m0 + tt;
            float dlt = delta[(size_t)m * EHH + c];
            float uv  = u[(size_t)m * EHH + c];
            const float* Bt = &sBC[tt * 256];
            const float* Ct = Bt + HH;

            float acc = 0.0f;
#pragma unroll
            for (int j = 0; j < 4; ++j) {
                int n = lane + 32 * j;
                float dA  = 0.5f * dlt * an[j];
                float inv = 1.0f / (1.0f - dA);
                h[j] = (1.0f + dA) * inv * h[j] + dlt * uv * Bt[n] * inv;
                acc += h[j] * Ct[n];
            }
#pragma unroll
            for (int off = 16; off; off >>= 1) acc += __shfl_xor(acc, off, 32);
            if (lane == 0) y[(size_t)m * EHH + c] = acc + uv * Dc;
        }
        __syncthreads();
    }
}

// ---------------------------------------------------------------------------
// y *= silu(skip);  also emit f16 for the out-proj WMMA
// ---------------------------------------------------------------------------
__global__ void k_gate(const float* __restrict__ xs, float* __restrict__ y,
                       _Float16* __restrict__ yh) {
    int idx = blockIdx.x * blockDim.x + threadIdx.x;
    if (idx >= MM * EHH) return;
    int m = idx / EHH, c = idx % EHH;
    float o = y[idx] * siluf(xs[(size_t)m * (2 * EHH) + EHH + c]);
    y[idx]  = o;
    yh[idx] = (_Float16)o;
}

__global__ void k_residual(float* __restrict__ xcur, const float* __restrict__ t, int n) {
    int i = blockIdx.x * blockDim.x + threadIdx.x;
    if (i < n) xcur[i] += t[i];
}

// ---------------------------------------------------------------------------
extern "C" void kernel_launch(void* const* d_in, const int* in_sizes, int n_in,
                              void* d_out, int out_size, void* d_ws, size_t ws_size,
                              hipStream_t stream) {
    (void)in_sizes; (void)n_in; (void)out_size; (void)ws_size;
    char* ws = (char*)d_ws;
    size_t off = 0;
    auto alloc = [&](size_t bytes) -> void* {
        off = (off + 255) & ~(size_t)255;
        void* p = ws + off;
        off += bytes;
        return p;
    };

    float*    xcur      = (float*)   alloc((size_t)MM * HH * 4);
    float*    tmp       = (float*)   alloc((size_t)MM * HH * 4);
    _Float16* xn_h      = (_Float16*)alloc((size_t)MM * HH * 2);
    _Float16* in_w_h    = (_Float16*)alloc((size_t)512 * 128 * 2);
    _Float16* param_w_h = (_Float16*)alloc((size_t)NPAD * 256 * 2);
    _Float16* out_w_h   = (_Float16*)alloc((size_t)128 * 256 * 2);
    float*    xs        = (float*)   alloc((size_t)MM * 512 * 4);
    float*    u         = (float*)   alloc((size_t)MM * EHH * 4);
    _Float16* u_h       = (_Float16*)alloc((size_t)MM * EHH * 2);
    float*    dbc       = (float*)   alloc((size_t)MM * NPAD * 4);
    float*    delta     = (float*)   alloc((size_t)MM * EHH * 4);
    float*    ybuf      = (float*)   alloc((size_t)MM * EHH * 4);
    _Float16* y_h       = (_Float16*)alloc((size_t)MM * EHH * 2);

    hipMemcpyAsync(xcur, d_in[0], (size_t)MM * HH * 4, hipMemcpyDeviceToDevice, stream);

    const int T = 256;
    for (int l = 0; l < 2; ++l) {
        int base = 1 + l * 12;
        const float* in_w    = (const float*)d_in[base + 0];
        const float* in_b    = (const float*)d_in[base + 1];
        const float* out_w   = (const float*)d_in[base + 2];
        const float* out_b   = (const float*)d_in[base + 3];
        const float* param_w = (const float*)d_in[base + 4];
        const float* dt_w    = (const float*)d_in[base + 5];
        const float* dt_b    = (const float*)d_in[base + 6];
        const float* conv_w  = (const float*)d_in[base + 7];
        const float* conv_b  = (const float*)d_in[base + 8];
        const float* A_log   = (const float*)d_in[base + 9];
        const float* Dvec    = (const float*)d_in[base + 10];
        const float* norm_w  = (const float*)d_in[base + 11];

        k_convert<<<(512 * 128 + T - 1) / T, T, 0, stream>>>(in_w, in_w_h, 512 * 128);
        k_convert_pad<<<(NPAD * 256 + T - 1) / T, T, 0, stream>>>(param_w, param_w_h, 264,
                                                                  NPAD, 256);
        k_convert<<<(128 * 256 + T - 1) / T, T, 0, stream>>>(out_w, out_w_h, 128 * 256);

        k_rmsnorm<<<MM / 8, T, 0, stream>>>(xcur, norm_w, xn_h, nullptr, MM);

        // in-proj: [1024,512] = xn_h[1024,128] * in_w[512,128]^T + in_b  (K=128, NBLK=4)
        {
            int waves = (MM / 16) * ((512 / 16) / 4);
            k_gemm_wmma<128, 4><<<(waves * 32 + T - 1) / T, T, 0, stream>>>(
                xn_h, in_w_h, in_b, xs, MM, 512, 512);
        }

        k_conv_silu<<<(MM * EHH + T - 1) / T, T, 0, stream>>>(xs, conv_w, conv_b, u, u_h);

        // param-proj: [1024,272] = u_h[1024,256] * param_w[272,256]^T  (K=256, NBLK=1)
        {
            int waves = (MM / 16) * (NPAD / 16);
            k_gemm_wmma<256, 1><<<(waves * 32 + T - 1) / T, T, 0, stream>>>(
                u_h, param_w_h, nullptr, dbc, MM, NPAD, NPAD);
        }

        k_delta<<<(MM * EHH + T - 1) / T, T, 0, stream>>>(dbc, dt_w, dt_b, delta);

        // SSM recurrence: 64 blocks x 8 waves, TDM-staged B/C
        k_scan<<<BB * EHH / 8, T, 0, stream>>>(delta, u, dbc, A_log, Dvec, ybuf);

        k_gate<<<(MM * EHH + T - 1) / T, T, 0, stream>>>(xs, ybuf, y_h);

        // out-proj: [1024,128] = y_h[1024,256] * out_w[128,256]^T + out_b  (K=256, NBLK=4)
        {
            int waves = (MM / 16) * ((HH / 16) / 4);
            k_gemm_wmma<256, 4><<<(waves * 32 + T - 1) / T, T, 0, stream>>>(
                y_h, out_w_h, out_b, tmp, MM, HH, HH);
        }

        k_residual<<<(MM * HH + T - 1) / T, T, 0, stream>>>(xcur, tmp, MM * HH);
    }

    k_rmsnorm<<<MM / 8, T, 0, stream>>>(xcur, (const float*)d_in[25], nullptr,
                                        (float*)d_out, MM);
}
